// StructuralCausalModel_54494545051775
// MI455X (gfx1250) — matrix operand order repaired and checked by reading, not compiled, not run on hardware
//
#include <hip/hip_runtime.h>
#include <stdint.h>

// Problem constants (match reference)
#define Dn 256
#define Hn 64
#define Bn 8192
#define BT 128              // batch rows per workgroup
#define ZSTR 264            // padded z row stride (elements) -> 528B, conflict-free b128 reads
#define WSTEP (Dn * Hn)     // bf16 elements of packed weights per step (16384 = 32KB)
#define SMEM_BYTES (BT * ZSTR * 2 + 2 * WSTEP * 2)   // 67584 + 65536 = 133120 B

typedef __attribute__((ext_vector_type(4)))  unsigned int v4u;
typedef __attribute__((ext_vector_type(4)))  int          v4i;
typedef __attribute__((ext_vector_type(8)))  float        v8f;
typedef __attribute__((ext_vector_type(16))) __bf16       v16bf;

union ABFrag { v4u u[2]; v16bf v; };

#if defined(__has_builtin)
#if __has_builtin(__builtin_amdgcn_global_load_async_to_lds_b128) && \
    __has_builtin(__builtin_amdgcn_s_wait_asynccnt)
#define SCM_ASYNC 1
#endif
#endif

// ---------------------------------------------------------------------------
// Kernel 1: fold adjacency mask into W1, convert to bf16, and store in the
// WMMA B-fragment layout this code uses (k contiguous per lane half):
//   ws[((i*8 + kb)*4 + nt)*512 + L*16 + t]  holds  Wm[k][n]
//   with k = kb*32 + (L>>4)*16 + t,  n = nt*16 + (L&15)
//   Wm[k][n] = W1[i][n][k] * sigmoid(adj_logits[k][i]) * (k != i)
// Each lane's 16 bf16 (32B) are contiguous -> 2x ds_load_b128 at use time.
// ---------------------------------------------------------------------------
__global__ __launch_bounds__(256) void scm_pack_weights(
    const float* __restrict__ W1, const float* __restrict__ adjl,
    __bf16* __restrict__ wpk) {
  // one thread per 16-element lane chunk: D*8*4*32 = 262144 chunks
  int c = blockIdx.x * 256 + threadIdx.x;
  int L  = c & 31;
  int nt = (c >> 5) & 3;
  int kb = (c >> 7) & 7;
  int i  = c >> 10;
  int n  = nt * 16 + (L & 15);
  int kbase = kb * 32 + ((L >> 4) << 4);

  __bf16 vals[16];
#pragma unroll
  for (int t = 0; t < 16; ++t) {
    int k = kbase + t;
    float w = W1[(size_t)i * (Hn * Dn) + (size_t)n * Dn + k];
    float a = 0.0f;
    if (k != i) {
      float lg = adjl[(size_t)k * Dn + i];
      a = 1.0f / (1.0f + __expf(-lg));
    }
    vals[t] = (__bf16)(w * a);
  }
  // 32B contiguous store
  v4u* dst = (v4u*)(wpk + (size_t)c * 16);
  dst[0] = *(const v4u*)&vals[0];
  dst[1] = *(const v4u*)&vals[8];
}

// ---------------------------------------------------------------------------
// Weight slab prefetch: step's packed weights (kbNeed * 4KB, kb-major and
// contiguous) from global (L2-resident) into an LDS double buffer.
// ---------------------------------------------------------------------------
__device__ __forceinline__ void prefetch_w(const __bf16* gsrc, __bf16* lds,
                                           int kbNeed, int tid) {
  const char* g = (const char*)gsrc;
  char* l = (char*)lds;
#ifdef SCM_ASYNC
  for (int r = 0; r < kbNeed; ++r) {
    int off = r * 4096 + tid * 16;
    __builtin_amdgcn_global_load_async_to_lds_b128(
        (__attribute__((address_space(1))) v4i*)(g + off),
        (__attribute__((address_space(3))) v4i*)(l + off), 0, 0);
  }
#else
  for (int r = 0; r < kbNeed; ++r) {
    int off = r * 4096 + tid * 16;
    *(v4u*)(l + off) = *(const v4u*)(g + off);
  }
#endif
}

// ---------------------------------------------------------------------------
// Kernel 2: sequential SCM scan. 8 waves; wave w owns batch rows
// [wgBase + 16w, +16). Per step i:
//   acc[16,64] += z_tile[16, 32k] @ Wm[i][32k, 64]   (bf16 WMMA, f32 acc)
//   det = silu(acc + b1) @ W2 ; val = det + b2 + sigma*noise ; z[:,i] = val
// Per K-block: all 10 ds_load_b128 (A + 4 B fragments, distinct registers)
// issue before the 4 WMMAs -> progressive partial s_wait_dscnt, no copies.
// ---------------------------------------------------------------------------
__global__ __launch_bounds__(256) void scm_scan(
    const float* __restrict__ noise, const float* __restrict__ b1,
    const float* __restrict__ w2, const float* __restrict__ b2,
    const float* __restrict__ lsig, const __bf16* __restrict__ wpk,
    float* __restrict__ out) {
  extern __shared__ char smem[];
  __bf16* zs    = (__bf16*)smem;                       // [BT][ZSTR] bf16
  __bf16* wbuf0 = (__bf16*)(smem + BT * ZSTR * 2);     // WSTEP bf16
  __bf16* wbuf1 = wbuf0 + WSTEP;

  const int tid  = threadIdx.x;
  const int wave = tid >> 5;
  const int lane = tid & 31;
  const int half = lane >> 4;       // A/B layout half-wave selector
  const int m16  = lane & 15;
  const int rowBase = blockIdx.x * BT;

  // zero z tile (z0 = 0); pad region included
  {
    uint32_t* zp = (uint32_t*)zs;
    const int n32 = (BT * ZSTR) / 2;
    for (int e = tid; e < n32; e += 256) zp[e] = 0u;
  }
  // step 0 consumes zero K-blocks, so no initial weight prefetch needed

  const __bf16* zrow = zs + (wave * 16 + m16) * ZSTR;

  for (int i = 0; i < Dn; ++i) {
#ifdef SCM_ASYNC
    __builtin_amdgcn_s_wait_asynccnt(0);   // my async slab writes landed
#endif
    __syncthreads();                       // slab complete + z col i-1 visible

    // prefetch next step's weight slab into the other buffer
    if (i + 1 < Dn) {
      int kbNeed = (i + 32) >> 5;          // ceil((i+1)/32)
      __bf16* dst = ((i + 1) & 1) ? wbuf1 : wbuf0;
      prefetch_w(wpk + (size_t)(i + 1) * WSTEP, dst, kbNeed, tid);
    }

    const __bf16* wb = (i & 1) ? wbuf1 : wbuf0;

    v8f acc[4] = {};                       // 4 N-tiles of [16,16] f32
    const int kB = (i + 31) >> 5;          // only nonzero z columns

    for (int kb = 0; kb < kB; ++kb) {
      const __bf16* za = zrow + kb * 32 + half * 16;
      const __bf16* wt = wb + kb * 2048 + lane * 16;
      ABFrag af, bf0, bf1, bf2, bf3;
      // issue all fragment loads first (fresh registers, in-order DS returns)
      af.u[0]  = *(const v4u*)(za);
      af.u[1]  = *(const v4u*)(za + 8);
      bf0.u[0] = *(const v4u*)(wt);
      bf0.u[1] = *(const v4u*)(wt + 8);
      bf1.u[0] = *(const v4u*)(wt + 512);
      bf1.u[1] = *(const v4u*)(wt + 520);
      bf2.u[0] = *(const v4u*)(wt + 1024);
      bf2.u[1] = *(const v4u*)(wt + 1032);
      bf3.u[0] = *(const v4u*)(wt + 1536);
      bf3.u[1] = *(const v4u*)(wt + 1544);
      // then consume: waits legalize progressively (0x6 -> 0x4 -> 0x2 -> 0x0)
      acc[0] = __builtin_amdgcn_wmma_f32_16x16x32_bf16(
          false, af.v, false, bf0.v, (short)0, acc[0], false, false);
      acc[1] = __builtin_amdgcn_wmma_f32_16x16x32_bf16(
          false, af.v, false, bf1.v, (short)0, acc[1], false, false);
      acc[2] = __builtin_amdgcn_wmma_f32_16x16x32_bf16(
          false, af.v, false, bf2.v, (short)0, acc[2], false, false);
      acc[3] = __builtin_amdgcn_wmma_f32_16x16x32_bf16(
          false, af.v, false, bf3.v, (short)0, acc[3], false, false);
    }

    // epilogue: silu, H-reduction against W2, noise injection
    float b1v[4], w2v[4];
#pragma unroll
    for (int nt = 0; nt < 4; ++nt) {
      int n = nt * 16 + m16;               // C layout: N = lane%16 per tile
      b1v[nt] = b1[i * Hn + n];
      w2v[nt] = w2[i * Hn + n];
    }
    const float b2i  = b2[i];
    const float sigi = __expf(lsig[i]);

#pragma unroll
    for (int j = 0; j < 8; ++j) {          // C VGPR j: M=j (lanes<16), M=j+8
      float s = 0.0f;
#pragma unroll
      for (int nt = 0; nt < 4; ++nt) {
        float x = acc[nt][j] + b1v[nt];
        float hx = x * (1.0f / (1.0f + __expf(-x)));   // silu
        s += hx * w2v[nt];
      }
      s += __shfl_xor(s, 1);
      s += __shfl_xor(s, 2);
      s += __shfl_xor(s, 4);
      s += __shfl_xor(s, 8);               // full sum within 16-lane group
      if (m16 == j) {
        int mloc = j + half * 8;
        int row  = rowBase + wave * 16 + mloc;
        float val = s + b2i + sigi * noise[(size_t)row * Dn + i];
        out[(size_t)row * Dn + i] = val;                  // f32 result
        zs[(wave * 16 + mloc) * ZSTR + i] = (__bf16)val;  // feed the chain
      }
    }
  }
}

extern "C" void kernel_launch(void* const* d_in, const int* in_sizes, int n_in,
                              void* d_out, int out_size, void* d_ws, size_t ws_size,
                              hipStream_t stream) {
  const float* noise = (const float*)d_in[0];
  const float* adjl  = (const float*)d_in[1];
  const float* W1    = (const float*)d_in[2];
  const float* b1    = (const float*)d_in[3];
  const float* W2    = (const float*)d_in[4];
  const float* b2    = (const float*)d_in[5];
  const float* lsig  = (const float*)d_in[6];
  __bf16* wpk = (__bf16*)d_ws;   // 8 MB packed masked weights

  // 1) fold adjacency into W1, bf16, WMMA-fragment layout
  scm_pack_weights<<<(Dn * 8 * 4 * 32) / 256, 256, 0, stream>>>(W1, adjl, wpk);
  // 2) sequential scan, batch-parallel across workgroups
  scm_scan<<<Bn / BT, 256, SMEM_BYTES, stream>>>(noise, b1, W2, b2, lsig, wpk,
                                                 (float*)d_out);
}
